// SelfAttention_73332271611949
// MI455X (gfx1250) — matrix operand order repaired and checked by reading, not compiled, not run on hardware
//
#include <hip/hip_runtime.h>
#include <hip/hip_bf16.h>

// MI455X (gfx1250) self-attention pipeline, all matmul stages on
// v_wmma_f32_16x16x32_f16 (wave32 WMMA, f32 accumulate), with async
// global->LDS tile staging (GLOBAL_LOAD_ASYNC_TO_LDS_B128 / ASYNCcnt)
// and double-buffered GEMM tiles.
//
// Stages (all on `stream`, sequential):
//   0) fp32 -> fp16 conversion of x / w_q / w_kv / w_out into d_ws
//   1) Q  = x @ w_q^T        (WMMA GEMM, f16 out)
//   2) KV = x @ w_kv^T       (WMMA GEMM, f16 out)
//   3) flash-style attention per (b, h, 128-row block): S = QK^T/8 kept in
//      registers, online softmax, O += P@V via WMMA, attn written f16
//   4) out = attn @ w_out^T + b_out  (WMMA GEMM, f32 out + bias)
//
// Workspace requirement: ~92.3 MB of d_ws (f16 intermediates).

typedef __attribute__((ext_vector_type(16))) _Float16 v16h;
typedef __attribute__((ext_vector_type(8)))  _Float16 v8h;
typedef __attribute__((ext_vector_type(4)))  _Float16 v4h;
typedef __attribute__((ext_vector_type(8)))  float    v8f;
typedef __attribute__((ext_vector_type(4)))  int      v4i;

#define B_   4
#define N_   2048
#define D_   1024
#define H_   16
#define E_   64
#define DH_  1024   // H_*E_

// ---------------------------------------------------------------------------
// Async global->LDS copy (gfx1250 GLOBAL_LOAD_ASYNC_TO_LDS_B128, ASYNCcnt),
// guarded; falls back to a synchronous 16-byte copy through VGPRs.
// Builtin signature (from toolchain diagnostic): arg0 = addrspace(1) v4i*
// (printed as "int4 __device__ *"), arg1 = LDS side, then imm offset, imm cpol.
// ---------------------------------------------------------------------------
#if defined(__has_builtin)
#if __has_builtin(__builtin_amdgcn_global_load_async_to_lds_b128) && \
    __has_builtin(__builtin_amdgcn_s_wait_asynccnt)
#define HAVE_ASYNC_LDS 1
#endif
#endif

typedef __attribute__((address_space(1))) v4i g_v4i;   // global b128 payload
typedef __attribute__((address_space(3))) v4i l_v4i;   // LDS b128 payload

__device__ __forceinline__ void copy16_g2l(_Float16* lds, const _Float16* g) {
#if defined(HAVE_ASYNC_LDS)
  __builtin_amdgcn_global_load_async_to_lds_b128((g_v4i*)g, (l_v4i*)lds, 0, 0);
#else
  *(v8h*)lds = *(const v8h*)g;
#endif
}

__device__ __forceinline__ void wait_g2l() {
#if defined(HAVE_ASYNC_LDS)
  __builtin_amdgcn_s_wait_asynccnt(0);
#endif
}

// ---------------------------------------------------------------------------
// fp32 -> fp16 conversion (vectorized by 4)
// ---------------------------------------------------------------------------
__global__ __launch_bounds__(256)
void cvt_f32_to_f16_kernel(const float* __restrict__ in,
                           _Float16* __restrict__ out, int n4) {
  int i = blockIdx.x * 256 + threadIdx.x;
  if (i < n4) {
    const float4 f = ((const float4*)in)[i];
    v4h hv = { (_Float16)f.x, (_Float16)f.y, (_Float16)f.z, (_Float16)f.w };
    ((v4h*)out)[i] = hv;
  }
}

// ---------------------------------------------------------------------------
// WMMA GEMM: C[M,Ncols] = A[M,K] * Bw[Ncols,K]^T   (Bw is torch-style [out,in])
// Workgroup tile 128x128, BK=32, 8 waves (2x4), each wave 64x32 = 8 frags.
// Double-buffered LDS; async staging of tile k+1 overlaps WMMAs on tile k.
// ---------------------------------------------------------------------------
template<bool F16OUT, bool BIAS>
__global__ __launch_bounds__(256)
void gemm_wmma_kernel(const _Float16* __restrict__ A,   // [M,K] row-major
                      const _Float16* __restrict__ Bw,  // [Ncols,K] row-major
                      const float* __restrict__ bias,   // [Ncols] (if BIAS)
                      void* __restrict__ Cout,          // [M,Ncols]
                      int M, int Ncols, int K) {
  __shared__ __attribute__((aligned(32))) _Float16 lA[2][128 * 32];
  __shared__ __attribute__((aligned(32))) _Float16 lB[2][128 * 32];

  const int t    = threadIdx.x;
  const int lane = t & 31;
  const int wave = t >> 5;
  const int wm   = wave >> 2;    // 0..1  (M direction)
  const int wn   = wave & 3;     // 0..3  (N direction)
  const int half = lane >> 4;    // lane group 0..15 / 16..31
  const int l16  = lane & 15;

  const int mBase = blockIdx.x * 128;
  const int nBase = blockIdx.y * 128;

  // staging: chunk per thread = 8 f16 (16B); 128 rows x 32 cols = 512 chunks
  const int r0 = t >> 2;          // rows 0..63 (pass0), +64 (pass1)
  const int c0 = (t & 3) * 8;     // col offset within 32

  const _Float16* gA0 = &A [(size_t)(mBase + r0     ) * K + c0];
  const _Float16* gA1 = &A [(size_t)(mBase + r0 + 64) * K + c0];
  const _Float16* gB0 = &Bw[(size_t)(nBase + r0     ) * K + c0];
  const _Float16* gB1 = &Bw[(size_t)(nBase + r0 + 64) * K + c0];

  v8f acc[4][2] = {};

  // prologue: stage tile k0=0 into buffer 0
  copy16_g2l(&lA[0][(r0     ) * 32 + c0], gA0);
  copy16_g2l(&lA[0][(r0 + 64) * 32 + c0], gA1);
  copy16_g2l(&lB[0][(r0     ) * 32 + c0], gB0);
  copy16_g2l(&lB[0][(r0 + 64) * 32 + c0], gB1);
  wait_g2l();
  __syncthreads();

  int buf = 0;
  for (int k0 = 0; k0 < K; k0 += 32, buf ^= 1) {
    // stage next tile into the other buffer (overlaps with compute below)
    if (k0 + 32 < K) {
      copy16_g2l(&lA[buf ^ 1][(r0     ) * 32 + c0], gA0 + k0 + 32);
      copy16_g2l(&lA[buf ^ 1][(r0 + 64) * 32 + c0], gA1 + k0 + 32);
      copy16_g2l(&lB[buf ^ 1][(r0     ) * 32 + c0], gB0 + k0 + 32);
      copy16_g2l(&lB[buf ^ 1][(r0 + 64) * 32 + c0], gB1 + k0 + 32);
    }
    if (k0 + 64 < K) {  // gfx1250 global_prefetch_b8 two tiles ahead
      __builtin_prefetch(gA0 + k0 + 64, 0, 1);
      __builtin_prefetch(gB0 + k0 + 64, 0, 1);
    }

    // B fragments: lane holds col n=l16, contiguous K (half selects K 0-15/16-31)
    v16h bfr[2];
#pragma unroll
    for (int nf = 0; nf < 2; nf++)
      bfr[nf] =
          *(const v16h*)&lB[buf][(wn * 32 + nf * 16 + l16) * 32 + half * 16];

#pragma unroll
    for (int mf = 0; mf < 4; mf++) {
      const int m = wm * 64 + mf * 16 + l16;
      // A layout: half=0 -> K{0..15}; half=1 -> K{16..31}
      v16h a = *(const v16h*)&lA[buf][m * 32 + half * 16];
#pragma unroll
      for (int nf = 0; nf < 2; nf++)
        acc[mf][nf] = __builtin_amdgcn_wmma_f32_16x16x32_f16(
            false, a, false, bfr[nf], (short)0, acc[mf][nf], false, false);
    }

    wait_g2l();       // next tile fully landed in LDS
    __syncthreads();  // all waves done reading current buffer
  }

  // C/D layout: lane half picks M rows half*8+v, col n = l16
#pragma unroll
  for (int mf = 0; mf < 4; mf++) {
#pragma unroll
    for (int nf = 0; nf < 2; nf++) {
      const int m0 = mBase + wm * 64 + mf * 16 + half * 8;
      const int n  = nBase + wn * 32 + nf * 16 + l16;
      const float bv = BIAS ? bias[n] : 0.0f;
#pragma unroll
      for (int v = 0; v < 8; v++) {
        const float val = acc[mf][nf][v] + bv;
        if (F16OUT)
          ((_Float16*)Cout)[(size_t)(m0 + v) * Ncols + n] = (_Float16)val;
        else
          ((float*)Cout)[(size_t)(m0 + v) * Ncols + n] = val;
      }
    }
  }
}

// ---------------------------------------------------------------------------
// Flash attention: one WG per (b, h, 128 i-rows). 8 waves, one 16-row strip
// each. j-blocks of 64. Q frags live in registers for the whole j loop.
// ---------------------------------------------------------------------------
__global__ __launch_bounds__(256)
void attn_wmma_kernel(const _Float16* __restrict__ Q,   // [B*N, DH]
                      const _Float16* __restrict__ KV,  // [B*N, 2*DH]
                      _Float16* __restrict__ O) {       // [B*N, DH]
  __shared__ __attribute__((aligned(32))) _Float16 lK [64 * 64];      // [j][e]
  __shared__ __attribute__((aligned(32))) _Float16 lVt[64 * 64];      // [e][j]
  __shared__ __attribute__((aligned(32))) _Float16 lP [8 * 16 * 64];  // per-wave [i][j]

  const int t    = threadIdx.x;
  const int lane = t & 31;
  const int wave = t >> 5;
  const int half = lane >> 4;
  const int l16  = lane & 15;
  const int h    = blockIdx.y;
  const int b    = blockIdx.z;
  const size_t rowQ0 = (size_t)b * N_ + blockIdx.x * 128;

  // Q fragments, pre-scaled by 1/sqrt(E)=0.125 (exact in f16)
  v16h qf[2];
  {
    const _Float16* qp = Q + (rowQ0 + wave * 16 + l16) * DH_ + h * E_;
#pragma unroll
    for (int ks = 0; ks < 2; ks++) {
      v16h a = *(const v16h*)(qp + ks * 32 + half * 16);
#pragma unroll
      for (int i = 0; i < 16; i++) a[i] *= (_Float16)0.125f;
      qf[ks] = a;
    }
  }

  float mrow[8], lrow[8];
  v8f o[4] = {};
#pragma unroll
  for (int v = 0; v < 8; v++) { mrow[v] = -3.0e38f; lrow[v] = 0.0f; }

  for (int jb = 0; jb < N_; jb += 64) {
    __syncthreads();
    {
      const size_t rb = (size_t)b * N_ + jb;
      // K tile [64][64]: async 16B chunks, coalesced
      const int kr = t >> 3, kc = (t & 7) * 8;
      copy16_g2l(&lK[kr * 64 + kc],
                 &KV[(rb + kr) * (2 * DH_) + h * E_ + kc]);
      copy16_g2l(&lK[(kr + 32) * 64 + kc],
                 &KV[(rb + kr + 32) * (2 * DH_) + h * E_ + kc]);
      // V tile transposed into [e][j] (scalar: no 16-bit async op)
#pragma unroll
      for (int i = 0; i < 16; i++) {
        const int idx = i * 256 + t;
        const int vj = idx >> 6, ve = idx & 63;
        lVt[ve * 64 + vj] = KV[(rb + vj) * (2 * DH_) + DH_ + h * E_ + ve];
      }
    }
    wait_g2l();
    __syncthreads();

    // S = (Q/8) @ K^T : 4 j-frags x 2 k-steps
    v8f s[4];
#pragma unroll
    for (int jf = 0; jf < 4; jf++) {
      v8f acc = {};
#pragma unroll
      for (int ks = 0; ks < 2; ks++) {
        v16h bf = *(const v16h*)&lK[(jf * 16 + l16) * 64 + ks * 32 + half * 16];
        acc = __builtin_amdgcn_wmma_f32_16x16x32_f16(
            false, qf[ks], false, bf, (short)0, acc, false, false);
      }
      s[jf] = acc;
    }

    // online softmax (rows M = half*8+v; reduce across the 16 N-lanes)
    float al[8];
#pragma unroll
    for (int v = 0; v < 8; v++) {
      float mx = s[0][v];
#pragma unroll
      for (int jf = 1; jf < 4; jf++) mx = fmaxf(mx, s[jf][v]);
#pragma unroll
      for (int msk = 1; msk < 16; msk <<= 1)
        mx = fmaxf(mx, __shfl_xor(mx, msk, 32));
      const float mn = fmaxf(mrow[v], mx);
      al[v] = __expf(mrow[v] - mn);
      mrow[v] = mn;
      float rs = 0.0f;
#pragma unroll
      for (int jf = 0; jf < 4; jf++) {
        const float p = __expf(s[jf][v] - mn);
        s[jf][v] = p;
        rs += p;
      }
#pragma unroll
      for (int msk = 1; msk < 16; msk <<= 1) rs += __shfl_xor(rs, msk, 32);
      lrow[v] = lrow[v] * al[v] + rs;
    }
#pragma unroll
    for (int ef = 0; ef < 4; ef++)
#pragma unroll
      for (int v = 0; v < 8; v++) o[ef][v] *= al[v];

    // write P strip (per-wave) to LDS in C-layout, reread in A-layout
    _Float16* myP = &lP[wave * 16 * 64];
#pragma unroll
    for (int jf = 0; jf < 4; jf++)
#pragma unroll
      for (int v = 0; v < 8; v++)
        myP[(half * 8 + v) * 64 + jf * 16 + l16] = (_Float16)s[jf][v];
    __syncthreads();

    // O += P @ V : 2 k-steps x 4 e-frags
#pragma unroll
    for (int ks = 0; ks < 2; ks++) {
      v16h pa = *(const v16h*)&myP[l16 * 64 + ks * 32 + half * 16];
#pragma unroll
      for (int ef = 0; ef < 4; ef++) {
        v16h vf = *(const v16h*)&lVt[(ef * 16 + l16) * 64 + ks * 32 + half * 16];
        o[ef] = __builtin_amdgcn_wmma_f32_16x16x32_f16(
            false, pa, false, vf, (short)0, o[ef], false, false);
      }
    }
  }

  // epilogue: O /= rowsum, store f16
#pragma unroll
  for (int v = 0; v < 8; v++) {
    const float inv = 1.0f / lrow[v];
    const size_t row = rowQ0 + wave * 16 + half * 8 + v;
#pragma unroll
    for (int ef = 0; ef < 4; ef++)
      O[row * DH_ + h * E_ + ef * 16 + l16] = (_Float16)(o[ef][v] * inv);
  }
}

// ---------------------------------------------------------------------------
extern "C" void kernel_launch(void* const* d_in, const int* in_sizes, int n_in,
                              void* d_out, int out_size, void* d_ws,
                              size_t ws_size, hipStream_t stream) {
  (void)in_sizes; (void)n_in; (void)out_size; (void)ws_size;
  const float* x     = (const float*)d_in[0];
  const float* w_q   = (const float*)d_in[1];
  const float* w_kv  = (const float*)d_in[2];
  const float* w_out = (const float*)d_in[3];
  const float* b_out = (const float*)d_in[4];

  // workspace carve-up (f16 elements); total ~92.3 MB
  _Float16* ws   = (_Float16*)d_ws;
  _Float16* xh   = ws;                                  // B*N*D   = 8388608
  _Float16* wqh  = xh   + (size_t)B_ * N_ * D_;         // DH*D    = 1048576
  _Float16* wkvh = wqh  + (size_t)DH_ * D_;             // 2*DH*D  = 2097152
  _Float16* woh  = wkvh + (size_t)2 * DH_ * D_;         // D*DH    = 1048576
  _Float16* qh   = woh  + (size_t)D_ * DH_;             // B*N*DH  = 8388608
  _Float16* kvh  = qh   + (size_t)B_ * N_ * DH_;        // B*N*2DH = 16777216
  _Float16* ah   = kvh  + (size_t)B_ * N_ * 2 * DH_;    // B*N*DH  = 8388608

  // 0) conversions
  cvt_f32_to_f16_kernel<<<(B_ * N_ * D_ / 4) / 256, 256, 0, stream>>>(
      x, xh, B_ * N_ * D_ / 4);
  cvt_f32_to_f16_kernel<<<(DH_ * D_ / 4) / 256, 256, 0, stream>>>(
      w_q, wqh, DH_ * D_ / 4);
  cvt_f32_to_f16_kernel<<<(2 * DH_ * D_ / 4) / 256, 256, 0, stream>>>(
      w_kv, wkvh, 2 * DH_ * D_ / 4);
  cvt_f32_to_f16_kernel<<<(D_ * DH_ / 4) / 256, 256, 0, stream>>>(
      w_out, woh, D_ * DH_ / 4);

  const int M = B_ * N_;  // 8192
  // 1) Q = x @ w_q^T
  gemm_wmma_kernel<true, false><<<dim3(M / 128, DH_ / 128), 256, 0, stream>>>(
      xh, wqh, nullptr, qh, M, DH_, D_);
  // 2) KV = x @ w_kv^T
  gemm_wmma_kernel<true, false><<<dim3(M / 128, 2 * DH_ / 128), 256, 0, stream>>>(
      xh, wkvh, nullptr, kvh, M, 2 * DH_, D_);
  // 3) attention
  attn_wmma_kernel<<<dim3(N_ / 128, H_, B_), 256, 0, stream>>>(qh, kvh, ah);
  // 4) out = attn @ w_out^T + b_out (f32)
  gemm_wmma_kernel<false, true><<<dim3(M / 128, D_ / 128), 256, 0, stream>>>(
      ah, woh, b_out, d_out, M, D_, DH_);
}
